// SiglipAttention_4896262718076
// MI455X (gfx1250) — matrix operand order repaired and checked by reading, not compiled, not run on hardware
//
#include <hip/hip_runtime.h>
#include <hip/hip_bf16.h>

#define DEV __device__ __forceinline__

typedef __attribute__((ext_vector_type(16))) _Float16 v16h;
typedef __attribute__((ext_vector_type(8)))  _Float16 v8h;
typedef __attribute__((ext_vector_type(8)))  float    v8f;
typedef __attribute__((ext_vector_type(4)))  float    v4f;
typedef __attribute__((ext_vector_type(4)))  unsigned int u32x4;
typedef __attribute__((ext_vector_type(8)))  int      i32x8;
typedef __attribute__((ext_vector_type(4)))  int      i32x4;

// Scheduling fence: nothing may be reordered across this point.
DEV void sched_fence() {
#if __has_builtin(__builtin_amdgcn_sched_barrier)
  __builtin_amdgcn_sched_barrier(0);
#endif
}

// ---------------------------------------------------------------------------
// Tensor Data Mover support (gfx1250). This toolchain exposes the 6-arg
// builtin: (u32x4 g0, i32x8 g1, i32x4 g2, i32x4 g3, i32x8 extra, i32 cpol).
#if defined(__gfx1250__) && __has_builtin(__builtin_amdgcn_tensor_load_to_lds)
#define HAVE_TDM 1
#else
#define HAVE_TDM 0
#endif

#if HAVE_TDM
DEV void tdm_wait() {
#if __has_builtin(__builtin_amdgcn_s_wait_tensorcnt)
  __builtin_amdgcn_s_wait_tensorcnt(0);
#else
  asm volatile("s_wait_tensorcnt 0x0" ::: "memory");
#endif
}

// Generic pointer to shared memory: low 32 bits are the LDS byte address
// (ISA 10.2).
DEV unsigned lds_byte_addr(const void* p) {
  return (unsigned)(unsigned long long)(uintptr_t)p;
}

// 2D tile DMA: tile_d0 halves per row, tile_d1 rows, row stride stride0 halves.
// D# layout per CDNA5 ISA. data_size=2 bytes.
DEV void tdm_load_2d(unsigned lds_addr, const void* gaddr, unsigned tile_d0,
                     unsigned tile_d1, unsigned long long stride0, bool pad,
                     unsigned pad_interval_code, unsigned pad_amount_code) {
  unsigned long long ga = (unsigned long long)(uintptr_t)gaddr;
  const unsigned td0 = 1u << 20, td1 = 1u << 20;  // huge tensor dims: no OOB clip
  u32x4 g0;
  g0[0] = 1u;  // count=1 (valid), is_restore=0, gather_mode=0
  g0[1] = lds_addr;
  g0[2] = (unsigned)(ga & 0xffffffffull);
  g0[3] = (unsigned)((ga >> 32) & 0x01ffffffull) | (2u << 30);  // type=2 "image"
  i32x8 g1;
  g1[0] = (int)((1u << 16) |                    // data_size = 1 -> 2 bytes
                (pad ? (1u << 20) : 0u) |       // pad_enable
                (pad_interval_code << 22) | (pad_amount_code << 25));
  g1[1] = (int)((td0 & 0xffffu) << 16);                          // tensor_dim0[15:0]
  g1[2] = (int)((td0 >> 16) | ((td1 & 0xffffu) << 16));          // td0 hi | td1 lo
  g1[3] = (int)((td1 >> 16) | ((tile_d0 & 0xffffu) << 16));      // td1 hi | tile_dim0
  g1[4] = (int)(tile_d1 & 0xffffu);                              // tile_dim1, tile_dim2=0
  g1[5] = (int)(unsigned)(stride0 & 0xffffffffull);              // dim0_stride[31:0]
  g1[6] = (int)(unsigned)((stride0 >> 32) & 0xffffull);          // dim0_stride[47:32]
  g1[7] = 0;
  i32x4 z4 = {0, 0, 0, 0};
  i32x8 z8 = {0, 0, 0, 0, 0, 0, 0, 0};
  __builtin_amdgcn_tensor_load_to_lds(g0, g1, z4, z4, z8, 0);
}
#endif  // HAVE_TDM

// ---------------------------------------------------------------------------
// WMMA operand fragment loaders (16x16x32 f16, wave32; this code's contiguous k order, same in A and B).
// A-matrix 16x32 f16: lane L<16 -> row L, K=0..15; lane>=16 -> row L-16,
// K=16..31.
DEV v16h load_a_frag(const _Float16* base, int stride) {
  const int lane = threadIdx.x & 31;
  const int r    = lane & 15;
  const int kb   = (lane < 16) ? 0 : 16;
  const _Float16* p = base + r * stride + kb;
  v8h lo = *(const v8h*)(p);
  v8h hi = *(const v8h*)(p + 8);
  v16h out;
#pragma unroll
  for (int i = 0; i < 8; ++i) { out[i] = lo[i]; out[i + 8] = hi[i]; }
  return out;
}

// B-matrix 32x16 f16: lane<16 -> column N=lane, K=0..15; lane>=16 -> K=16..31.
// base points at (k=0,n=0) of a tile stored column-of-B contiguous ([N][K]).
DEV v16h load_b_frag(const _Float16* base, int stride) {
  const int lane = threadIdx.x & 31;
  const int n    = lane & 15;
  const int kb   = (lane < 16) ? 0 : 16;
  return *(const v16h*)(base + n * stride + kb);
}

// ---------------------------------------------------------------------------
static constexpr int Bc = 8;
static constexpr int Sc = 1024;
static constexpr int Dc = 768;
static constexpr int Hc = 12;
static constexpr int HDc = 64;
static constexpr int Mc = Bc * Sc;  // 8192

// ---------------------------------------------------------------------------
// Kernel 1: fused QKV projection, register-pipelined + double-buffered LDS.
// X[8192,768] fp32 x W[768,768] fp32 (+bias) -> f16.
// Q,K: [B,H,S,HD]; V transposed: [B,H,HD,S].
__global__ __launch_bounds__(256) void qkv_gemm_kernel(
    const float* __restrict__ X,
    const float* __restrict__ Wq, const float* __restrict__ bq,
    const float* __restrict__ Wk, const float* __restrict__ bk,
    const float* __restrict__ Wv, const float* __restrict__ bv,
    _Float16* __restrict__ Qo, _Float16* __restrict__ Ko,
    _Float16* __restrict__ Vo) {
  constexpr int LDA = 40;       // halves: 80B rows, 16B aligned, spreads banks
  constexpr int LDB = 48;       // halves: 96B rows, 32B aligned
  constexpr int ABUF = 128 * LDA;
  constexpr int BBUF = 64 * LDB;
  constexpr int NT = Dc / 32;   // 24 k-tiles
  __shared__ __attribute__((aligned(128))) _Float16 lds_a[2 * ABUF];
  __shared__ __attribute__((aligned(128))) _Float16 lds_b[2 * BBUF];

  const int mode = blockIdx.z;  // 0=Q 1=K 2=V
  const float* W    = (mode == 0) ? Wq : ((mode == 1) ? Wk : Wv);
  const float* bias = (mode == 0) ? bq : ((mode == 1) ? bk : bv);

  const int m0   = blockIdx.x * 128;
  const int n0   = blockIdx.y * 64;
  const int tid  = threadIdx.x;
  const int lane = tid & 31;
  const int wave = tid >> 5;
  const int wm   = (wave >> 1) * 32;
  const int wn   = (wave & 1) * 32;

  v8f c[2][2] = {};
  v4f ra[4], rb[2];

  auto load_regs = [&](int kb) {
#pragma unroll
    for (int it = 0; it < 4; ++it) {       // A: 128x32 fp32 = 1024 float4
      int slot = tid + it * 256;
      int row = slot >> 3, col = (slot & 7) * 4;
      ra[it] = *(const v4f*)(X + (size_t)(m0 + row) * Dc + kb + col);
    }
#pragma unroll
    for (int it = 0; it < 2; ++it) {       // B: 32x64 fp32 = 512 float4
      int slot = tid + it * 256;
      int k = slot >> 4, n = (slot & 15) * 4;
      rb[it] = *(const v4f*)(W + (size_t)(kb + k) * Dc + n0 + n);
    }
  };
  auto store_lds = [&](int buf) {
    _Float16* la = lds_a + buf * ABUF;
    _Float16* lb = lds_b + buf * BBUF;
#pragma unroll
    for (int it = 0; it < 4; ++it) {
      int slot = tid + it * 256;
      int row = slot >> 3, col = (slot & 7) * 4;
      _Float16* d = la + row * LDA + col;
      d[0] = (_Float16)ra[it].x; d[1] = (_Float16)ra[it].y;
      d[2] = (_Float16)ra[it].z; d[3] = (_Float16)ra[it].w;
    }
#pragma unroll
    for (int it = 0; it < 2; ++it) {       // transpose B into [n][k]
      int slot = tid + it * 256;
      int k = slot >> 4, n = (slot & 15) * 4;
      lb[(n + 0) * LDB + k] = (_Float16)rb[it].x;
      lb[(n + 1) * LDB + k] = (_Float16)rb[it].y;
      lb[(n + 2) * LDB + k] = (_Float16)rb[it].z;
      lb[(n + 3) * LDB + k] = (_Float16)rb[it].w;
    }
  };

  load_regs(0);
  store_lds(0);
  for (int i = 0; i < NT; ++i) {
    __syncthreads();                       // buffer i ready
    if (i + 1 < NT) load_regs((i + 1) * 32);   // issue all 6 b128 loads now
    sched_fence();                         // loads stay above; fly during WMMAs
    const int buf = i & 1;
    const _Float16* la = lds_a + buf * ABUF;
    const _Float16* lb = lds_b + buf * BBUF;
    v16h a0 = load_a_frag(la + (wm + 0)  * LDA, LDA);
    v16h a1 = load_a_frag(la + (wm + 16) * LDA, LDA);
    v16h b0 = load_b_frag(lb + (wn + 0)  * LDB, LDB);
    v16h b1 = load_b_frag(lb + (wn + 16) * LDB, LDB);
    c[0][0] = __builtin_amdgcn_wmma_f32_16x16x32_f16(false, a0, false, b0, (short)0, c[0][0], false, false);
    c[0][1] = __builtin_amdgcn_wmma_f32_16x16x32_f16(false, a0, false, b1, (short)0, c[0][1], false, false);
    c[1][0] = __builtin_amdgcn_wmma_f32_16x16x32_f16(false, a1, false, b0, (short)0, c[1][0], false, false);
    c[1][1] = __builtin_amdgcn_wmma_f32_16x16x32_f16(false, a1, false, b1, (short)0, c[1][1], false, false);
    sched_fence();                         // conversions may not hoist above
    if (i + 1 < NT) store_lds((i + 1) & 1);
  }

#pragma unroll
  for (int tm = 0; tm < 2; ++tm)
#pragma unroll
    for (int tn = 0; tn < 2; ++tn) {
#pragma unroll
      for (int i = 0; i < 8; ++i) {
        int row = i + ((lane < 16) ? 0 : 8);
        int col = lane & 15;
        int m = m0 + wm + tm * 16 + row;
        int n = n0 + wn + tn * 16 + col;
        float v = c[tm][tn][i] + bias[n];
        int b = m >> 10, s = m & 1023;
        int h = n >> 6,  hd = n & 63;
        if (mode < 2) {
          _Float16* dst = (mode == 0) ? Qo : Ko;
          dst[(((size_t)(b * Hc + h)) * Sc + s) * HDc + hd] = (_Float16)v;
        } else {
          Vo[(((size_t)(b * Hc + h)) * HDc + hd) * Sc + s] = (_Float16)v;
        }
      }
    }
}

// ---------------------------------------------------------------------------
// Kernel 2: flash attention. 8 waves/block share one (b,h); K/V 32-key chunks
// staged in double-buffered LDS by the Tensor Data Mover (wave 0 issues DMA).
__global__ __launch_bounds__(256) void attn_kernel(
    const _Float16* __restrict__ Q, const _Float16* __restrict__ K,
    const _Float16* __restrict__ V, _Float16* __restrict__ CTX) {
  __shared__ __attribute__((aligned(128))) _Float16 ldsK[2][32 * 64];  // [key][hd]
  __shared__ __attribute__((aligned(128))) _Float16 ldsV[2][64 * 32];  // [hd][key]
  __shared__ __attribute__((aligned(128))) _Float16 lds_p[8][16 * 32];

  const int wave = threadIdx.x >> 5;
  const int lane = threadIdx.x & 31;
  const int tid  = threadIdx.x;
  const int bh   = blockIdx.x >> 3;
  const int q0   = (blockIdx.x & 7) * 128 + wave * 16;
  const float scale = 0.125f;  // 1/sqrt(64)

  const _Float16* Qb = Q + ((size_t)bh * Sc + q0) * HDc;
  const _Float16* Kb = K + (size_t)bh * Sc * HDc;
  const _Float16* Vb = V + (size_t)bh * HDc * Sc;

  auto stage_kv = [&](int key0, int buf) {
#if HAVE_TDM
    if (wave == 0) {
      // K chunk: 32 rows x 64 halves, contiguous 4KB -> flat 1D DMA
      tdm_load_2d(lds_byte_addr(&ldsK[buf][0]), Kb + (size_t)key0 * HDc,
                  2048u, 0u, 2048ull, false, 0, 0);
      // V chunk: 64 rows of 32 halves, global row stride S halves
      tdm_load_2d(lds_byte_addr(&ldsV[buf][0]), Vb + key0,
                  32u, 64u, (unsigned long long)Sc, false, 0, 0);
    }
#else
    {  // fallback: all 256 threads copy 16B each
      v8h kk = *(const v8h*)(Kb + (size_t)key0 * HDc + tid * 8);
      *(v8h*)(&ldsK[buf][tid * 8]) = kk;
      int row = tid >> 2, part = tid & 3;
      v8h vv = *(const v8h*)(Vb + (size_t)row * Sc + key0 + part * 8);
      *(v8h*)(&ldsV[buf][row * 32 + part * 8]) = vv;
    }
#endif
  };

  v16h aq0 = load_a_frag(Qb + 0,  HDc);
  v16h aq1 = load_a_frag(Qb + 32, HDc);

  v8f o[4] = {};
  float mrow[8], lrow[8];
#pragma unroll
  for (int i = 0; i < 8; ++i) { mrow[i] = -1e30f; lrow[i] = 0.0f; }

  _Float16* P = &lds_p[wave][0];

  stage_kv(0, 0);
  for (int chunk = 0; chunk < Sc / 32; ++chunk) {
    const int key0 = chunk * 32;
    const int buf  = chunk & 1;
#if HAVE_TDM
    if (wave == 0) tdm_wait();            // chunk's DMA complete (wave0 issued)
#endif
    __syncthreads();                      // LDS K/V chunk visible to all waves
    if (chunk + 1 < Sc / 32) stage_kv(key0 + 32, buf ^ 1);  // overlap next DMA

    const _Float16* lk = &ldsK[buf][0];
    const _Float16* lv = &ldsV[buf][0];

    v16h bk00 = load_b_frag(lk + 0 * 64 + 0,   64);
    v16h bk01 = load_b_frag(lk + 0 * 64 + 32,  64);
    v16h bk10 = load_b_frag(lk + 16 * 64 + 0,  64);
    v16h bk11 = load_b_frag(lk + 16 * 64 + 32, 64);
    v8f c0 = {}, c1 = {};
    c0 = __builtin_amdgcn_wmma_f32_16x16x32_f16(false, aq0, false, bk00, (short)0, c0, false, false);
    c0 = __builtin_amdgcn_wmma_f32_16x16x32_f16(false, aq1, false, bk01, (short)0, c0, false, false);
    c1 = __builtin_amdgcn_wmma_f32_16x16x32_f16(false, aq0, false, bk10, (short)0, c1, false, false);
    c1 = __builtin_amdgcn_wmma_f32_16x16x32_f16(false, aq1, false, bk11, (short)0, c1, false, false);

#pragma unroll
    for (int i = 0; i < 8; ++i) {
      float s0 = c0[i] * scale;
      float s1 = c1[i] * scale;
      float mx = fmaxf(s0, s1);
#pragma unroll
      for (int off = 1; off < 16; off <<= 1) mx = fmaxf(mx, __shfl_xor(mx, off, 32));
      float mnew = fmaxf(mrow[i], mx);
      float corr = __expf(mrow[i] - mnew);
      float p0 = __expf(s0 - mnew);
      float p1 = __expf(s1 - mnew);
      float rs = p0 + p1;
#pragma unroll
      for (int off = 1; off < 16; off <<= 1) rs += __shfl_xor(rs, off, 32);
      lrow[i] = lrow[i] * corr + rs;
      mrow[i] = mnew;
#pragma unroll
      for (int t = 0; t < 4; ++t) o[t][i] *= corr;
      int row = i + ((lane < 16) ? 0 : 8);
      P[row * 32 + (lane & 15)]      = (_Float16)p0;
      P[row * 32 + 16 + (lane & 15)] = (_Float16)p1;
    }

    v16h ap = load_a_frag(P, 32);         // same-wave DS ops are in-order
#pragma unroll
    for (int t = 0; t < 4; ++t) {
      v16h bv = load_b_frag(lv + (t * 16) * 32, 32);
      o[t] = __builtin_amdgcn_wmma_f32_16x16x32_f16(false, ap, false, bv, (short)0, o[t], false, false);
    }
  }

  const int b = bh / Hc, h = bh % Hc;
#pragma unroll
  for (int t = 0; t < 4; ++t) {
#pragma unroll
    for (int i = 0; i < 8; ++i) {
      int row  = i + ((lane < 16) ? 0 : 8);
      int srow = q0 + row;
      float v = o[t][i] / lrow[i];
      CTX[((size_t)(b * Sc + srow)) * Dc + h * HDc + t * 16 + (lane & 15)] = (_Float16)v;
    }
  }
}

// ---------------------------------------------------------------------------
// Kernel 3: output projection. CTX f16 [8192,768] x Wo fp32 + bo -> fp32.
// A tile (already f16) staged by TDM with pad_enable to get the padded LDS
// layout; B staged through registers. Double-buffered.
__global__ __launch_bounds__(256) void out_gemm_kernel(
    const _Float16* __restrict__ CTX, const float* __restrict__ Wo,
    const float* __restrict__ bo, float* __restrict__ Out) {
  constexpr int LDA = 40;
  constexpr int LDB = 48;
  constexpr int ABUF = 128 * LDA;
  constexpr int BBUF = 64 * LDB;
  constexpr int NT = Dc / 32;
  __shared__ __attribute__((aligned(128))) _Float16 lds_a[2 * ABUF];
  __shared__ __attribute__((aligned(128))) _Float16 lds_b[2 * BBUF];

  const int m0   = blockIdx.x * 128;
  const int n0   = blockIdx.y * 64;
  const int tid  = threadIdx.x;
  const int lane = tid & 31;
  const int wave = tid >> 5;
  const int wm   = (wave >> 1) * 32;
  const int wn   = (wave & 1) * 32;

  v8f c[2][2] = {};
  v4f rb[2];
#if !HAVE_TDM
  v8h ra[2];
#endif

  auto load_phase = [&](int kb, int buf) {
#if HAVE_TDM
    if (wave == 0) {
      // A: 128 rows x 32 halves, global stride Dc halves; LDS rows padded
      // 32 halves + 8 halves: pad_interval 16 DWORDs (code 3), amount 4 (code 3)
      tdm_load_2d(lds_byte_addr(&lds_a[buf * ABUF]),
                  CTX + (size_t)m0 * Dc + kb, 32u, 128u,
                  (unsigned long long)Dc, true, 3, 3);
    }
#else
#pragma unroll
    for (int it = 0; it < 2; ++it) {
      int slot = tid + it * 256;
      int row = slot >> 2, col = (slot & 3) * 8;
      ra[it] = *(const v8h*)(CTX + (size_t)(m0 + row) * Dc + kb + col);
    }
#endif
#pragma unroll
    for (int it = 0; it < 2; ++it) {
      int slot = tid + it * 256;
      int k = slot >> 4, n = (slot & 15) * 4;
      rb[it] = *(const v4f*)(Wo + (size_t)(kb + k) * Dc + n0 + n);
    }
  };
  auto store_phase = [&](int buf) {
#if !HAVE_TDM
    _Float16* la = lds_a + buf * ABUF;
#pragma unroll
    for (int it = 0; it < 2; ++it) {
      int slot = tid + it * 256;
      int row = slot >> 2, col = (slot & 3) * 8;
      *(v8h*)(la + row * LDA + col) = ra[it];
    }
#endif
    _Float16* lb = lds_b + buf * BBUF;
#pragma unroll
    for (int it = 0; it < 2; ++it) {
      int slot = tid + it * 256;
      int k = slot >> 4, n = (slot & 15) * 4;
      lb[(n + 0) * LDB + k] = (_Float16)rb[it].x;
      lb[(n + 1) * LDB + k] = (_Float16)rb[it].y;
      lb[(n + 2) * LDB + k] = (_Float16)rb[it].z;
      lb[(n + 3) * LDB + k] = (_Float16)rb[it].w;
    }
  };

  load_phase(0, 0);
  store_phase(0);
  for (int i = 0; i < NT; ++i) {
#if HAVE_TDM
    if (wave == 0) tdm_wait();
#endif
    __syncthreads();
    if (i + 1 < NT) load_phase((i + 1) * 32, (i + 1) & 1);
    sched_fence();                        // keep loads/DMA issued up here
    const int buf = i & 1;
    const _Float16* la = lds_a + buf * ABUF;
    const _Float16* lb = lds_b + buf * BBUF;
    v16h a0 = load_a_frag(la + (wm + 0)  * LDA, LDA);
    v16h a1 = load_a_frag(la + (wm + 16) * LDA, LDA);
    v16h b0 = load_b_frag(lb + (wn + 0)  * LDB, LDB);
    v16h b1 = load_b_frag(lb + (wn + 16) * LDB, LDB);
    c[0][0] = __builtin_amdgcn_wmma_f32_16x16x32_f16(false, a0, false, b0, (short)0, c[0][0], false, false);
    c[0][1] = __builtin_amdgcn_wmma_f32_16x16x32_f16(false, a0, false, b1, (short)0, c[0][1], false, false);
    c[1][0] = __builtin_amdgcn_wmma_f32_16x16x32_f16(false, a1, false, b0, (short)0, c[1][0], false, false);
    c[1][1] = __builtin_amdgcn_wmma_f32_16x16x32_f16(false, a1, false, b1, (short)0, c[1][1], false, false);
    sched_fence();                        // conversions stay below the WMMAs
    if (i + 1 < NT) store_phase((i + 1) & 1);
  }

#pragma unroll
  for (int tm = 0; tm < 2; ++tm)
#pragma unroll
    for (int tn = 0; tn < 2; ++tn) {
#pragma unroll
      for (int i = 0; i < 8; ++i) {
        int row = i + ((lane < 16) ? 0 : 8);
        int col = lane & 15;
        int m = m0 + wm + tm * 16 + row;
        int n = n0 + wn + tn * 16 + col;
        Out[(size_t)m * Dc + n] = c[tm][tn][i] + bo[n];
      }
    }
}

// ---------------------------------------------------------------------------
extern "C" void kernel_launch(void* const* d_in, const int* in_sizes, int n_in,
                              void* d_out, int out_size, void* d_ws,
                              size_t ws_size, hipStream_t stream) {
  const float* X  = (const float*)d_in[0];
  const float* Wq = (const float*)d_in[1];
  const float* bq = (const float*)d_in[2];
  const float* Wk = (const float*)d_in[3];
  const float* bk = (const float*)d_in[4];
  const float* Wv = (const float*)d_in[5];
  const float* bv = (const float*)d_in[6];
  const float* Wo = (const float*)d_in[7];
  const float* bo = (const float*)d_in[8];
  float* out = (float*)d_out;

  const size_t per = (size_t)Mc * Dc;  // halves per tensor
  _Float16* ws = (_Float16*)d_ws;
  _Float16* Qw = ws;
  _Float16* Kw = ws + per;
  _Float16* Vw = ws + 2 * per;
  _Float16* Cw = ws + 3 * per;

  qkv_gemm_kernel<<<dim3(Mc / 128, Dc / 64, 3), 256, 0, stream>>>(
      X, Wq, bq, Wk, bk, Wv, bv, Qw, Kw, Vw);
  attn_kernel<<<dim3((Bc * Hc) * (Sc / 128)), 256, 0, stream>>>(Qw, Kw, Vw, Cw);
  out_gemm_kernel<<<dim3(Mc / 128, Dc / 64), 256, 0, stream>>>(Cw, Wo, bo, out);
}
